// Node_update_42528766165480
// MI455X (gfx1250) — matrix operand order repaired and checked by reading, not compiled, not run on hardware
//
#include <hip/hip_runtime.h>

// ---------------------------------------------------------------------------
// Node update (AlphaFold-style MSA row/col gated attention + FF) for gfx1250.
// B=1, L=1024, D=256, H=8, DH=32, D_PAIR=128.  All matmuls via
// v_wmma_f32_16x16x32_bf16 (wave32), GEMM tiles staged with the Tensor Data
// Mover (tensor_load_to_lds + s_wait_tensorcnt), fused epilogues.
// ---------------------------------------------------------------------------

#define SEQ   1024
#define DMSA  256
#define NH    8
#define DHH   32
#define DPAIR 128

typedef __bf16 bf16_t;
typedef __attribute__((ext_vector_type(16))) __bf16 bf16x16;
typedef __attribute__((ext_vector_type(8)))  __bf16 bf16x8;
typedef __attribute__((ext_vector_type(8)))  float  f32x8;
typedef __attribute__((ext_vector_type(4)))  unsigned int u32x4;
typedef __attribute__((ext_vector_type(8)))  int i32x8;
typedef __attribute__((ext_vector_type(4)))  int i32x4;

#if defined(__HIP_DEVICE_COMPILE__)
  #if __has_builtin(__builtin_amdgcn_tensor_load_to_lds) && __has_builtin(__builtin_amdgcn_s_wait_tensorcnt)
    #define USE_TDM 1
  #else
    #define USE_TDM 0
  #endif
#else
  #define USE_TDM 0
#endif

static __device__ __forceinline__ f32x8 zero8() {
  f32x8 z = {0.f, 0.f, 0.f, 0.f, 0.f, 0.f, 0.f, 0.f};
  return z;
}

// D = A(16x32 bf16) * B(32x16 bf16) + C(16x16 f32)
static __device__ __forceinline__ f32x8 wmma_bf16(bf16x16 a, bf16x16 b, f32x8 c) {
#if defined(__HIP_DEVICE_COMPILE__)
  return __builtin_amdgcn_wmma_f32_16x16x32_bf16(false, a, false, b, (short)0, c, false, false);
#else
  (void)a; (void)b;
  return c;
#endif
}

// A-matrix fragment (16x32, 16-bit): lane = row (lane&15); elems 0-7 = K k0..k0+7,
// elems 8-15 = K k0+16..k0+23, k0 = (lane<16) ? 0 : 8.  rowbase = &row[K=0 of chunk].
static __device__ __forceinline__ bf16x16 frag_a(const bf16_t* rowbase, int lane) {
  const bf16_t* p = rowbase + ((lane & 16) ? 8 : 0);
  bf16x8 lo = *(const bf16x8*)(p);
  bf16x8 hi = *(const bf16x8*)(p + 16);
  bf16x16 r;
#pragma unroll
  for (int i = 0; i < 8; ++i) { r[i] = lo[i]; r[i + 8] = hi[i]; }
  return r;
}

// B-matrix fragment (32x16, 16-bit): lane = col (lane&15); elems 0-7 = K k0..k0+7,
// elems 8-15 = K k0+16..k0+23, k0 = (lane<16) ? 0 : 8.  colbase points at K-contiguous storage of that column.
static __device__ __forceinline__ bf16x16 frag_b(const bf16_t* colbase, int lane) {
  const bf16_t* p = colbase + ((lane & 16) ? 8 : 0);
  bf16x8 lo = *(const bf16x8*)(p);
  bf16x8 hi = *(const bf16x8*)(p + 16);
  bf16x16 r;
#pragma unroll
  for (int i = 0; i < 8; ++i) { r[i] = lo[i]; r[i + 8] = hi[i]; }
  return r;
}

#if USE_TDM
// 2D tile load: tile_h rows x tile_w elems (2B each), row stride = stride elems,
// into LDS at byte offset lds_off (row-major, contiguous).  D# per ISA 08 §8.
static __device__ __forceinline__ void tdm_load_tile2d(unsigned int lds_off,
                                                       const bf16_t* gptr,
                                                       unsigned int tile_w,
                                                       unsigned int tile_h,
                                                       unsigned int stride_elems) {
  unsigned long long ga = (unsigned long long)(size_t)gptr;
  u32x4 g0;
  g0[0] = 1u;                                   // count=1, user descriptor
  g0[1] = lds_off;                              // lds_addr (bytes)
  g0[2] = (unsigned int)ga;                     // global_addr[31:0]
  g0[3] = (unsigned int)((ga >> 32) & 0x1FFFFFFull) | (2u << 30);  // addr[56:32] | type=2
  const unsigned int td0 = 1u << 20, td1 = 1u << 20;  // in-bounds tiles only
  i32x8 g1;
  g1[0] = (int)(1u << 16);                                          // data_size = 2B
  g1[1] = (int)((td0 & 0xFFFFu) << 16);                             // tensor_dim0 lo16
  g1[2] = (int)(((td0 >> 16) & 0xFFFFu) | ((td1 & 0xFFFFu) << 16)); // td0 hi | td1 lo
  g1[3] = (int)(((td1 >> 16) & 0xFFFFu) | (tile_w << 16));          // td1 hi | tile_dim0
  g1[4] = (int)(tile_h & 0xFFFFu);                                  // tile_dim1 (tile_dim2=0)
  g1[5] = (int)stride_elems;                                        // tensor_dim0_stride lo32
  g1[6] = 0;
  g1[7] = 0;
  i32x4 z4 = {0, 0, 0, 0};
  i32x8 z8 = {0, 0, 0, 0, 0, 0, 0, 0};
  __builtin_amdgcn_tensor_load_to_lds(g0, g1, z4, z4, z8, 0);
}
#endif

// ---------------------------------------------------------------------------
// Weight convert + transpose: W (K x N f32, row-major) -> WT (N x K bf16)
// so WMMA B-fragments read K-contiguous halves.
// ---------------------------------------------------------------------------
__global__ __launch_bounds__(256) void k_wT(const float* __restrict__ W,
                                            bf16_t* __restrict__ WT) {
  const int i = blockIdx.x * 256 + threadIdx.x;   // i = n*256 + k
  const int n = i >> 8, k = i & 255;
  WT[i] = (bf16_t)W[k * 256 + n];
}

// Wb (128 x 8 f32) -> WbT (16 x 128 bf16, cols 8..15 zero)
__global__ __launch_bounds__(256) void k_wbT(const float* __restrict__ Wb,
                                             bf16_t* __restrict__ WbT) {
  const int i = blockIdx.x * 256 + threadIdx.x;   // 2048
  const int h = i >> 7, c = i & 127;
  WbT[i] = (h < 8) ? (bf16_t)Wb[c * 8 + h] : (bf16_t)0.f;
}

// ---------------------------------------------------------------------------
// Row LayerNorm over 256 channels; optional f32 and bf16 outputs.
// ---------------------------------------------------------------------------
__global__ __launch_bounds__(256) void k_ln(const float* __restrict__ x,
                                            const float* __restrict__ g,
                                            const float* __restrict__ b,
                                            float* __restrict__ outf,
                                            bf16_t* __restrict__ outh) {
  __shared__ float red[16];
  const int row = blockIdx.x, t = threadIdx.x, lane = t & 31, wave = t >> 5;
  const float v = x[(size_t)row * DMSA + t];
  float s = v, s2 = v * v;
#pragma unroll
  for (int o = 16; o > 0; o >>= 1) { s += __shfl_xor(s, o, 32); s2 += __shfl_xor(s2, o, 32); }
  if (lane == 0) { red[wave] = s; red[8 + wave] = s2; }
  __syncthreads();
  if (t == 0) {
    float S = 0.f, S2 = 0.f;
#pragma unroll
    for (int w = 0; w < 8; ++w) { S += red[w]; S2 += red[8 + w]; }
    red[0] = S; red[1] = S2;
  }
  __syncthreads();
  const float mean = red[0] * (1.f / DMSA);
  const float var  = red[1] * (1.f / DMSA) - mean * mean;
  const float y = (v - mean) * rsqrtf(var + 1e-5f) * g[t] + b[t];
  if (outf) outf[(size_t)row * DMSA + t] = y;
  if (outh) outh[(size_t)row * DMSA + t] = (bf16_t)y;
}

// ---------------------------------------------------------------------------
// Pair bias: bias[q][k][h] = LN_c(pair[q,k,:]) @ Wb.  One wave per 16-key tile:
// LN in VALU, then 4 x wmma (K=128 in 32-chunks) against WbT (cols 8..15 = 0).
// This is the 512 MB streaming pass — pair is touched exactly once.
// ---------------------------------------------------------------------------
__global__ __launch_bounds__(128) void k_pair_bias(const float* __restrict__ pair,
                                                   const float* __restrict__ g,
                                                   const float* __restrict__ b,
                                                   const bf16_t* __restrict__ WbT,
                                                   float* __restrict__ bias) {
  __shared__ bf16_t sN[4][16 * DPAIR];
  const int tid = threadIdx.x, lane = tid & 31, wave = tid >> 5;
  const int q = blockIdx.x;
  const int kb = blockIdx.y * 64 + wave * 16;
  const int row = lane & 15, half = lane >> 4;       // 2 lanes per row, 64 ch each
  const float* pr = pair + ((size_t)q * SEQ + (kb + row)) * DPAIR + half * 64;
  float4 va[16];
#pragma unroll
  for (int i = 0; i < 16; ++i) va[i] = *(const float4*)(pr + i * 4);
  float s = 0.f, s2 = 0.f;
#pragma unroll
  for (int i = 0; i < 16; ++i) {
    s  += va[i].x + va[i].y + va[i].z + va[i].w;
    s2 += va[i].x * va[i].x + va[i].y * va[i].y + va[i].z * va[i].z + va[i].w * va[i].w;
  }
  s += __shfl_xor(s, 16, 32);
  s2 += __shfl_xor(s2, 16, 32);
  const float mean = s * (1.f / DPAIR);
  const float var  = s2 * (1.f / DPAIR) - mean * mean;
  const float rstd = rsqrtf(var + 1e-5f);
#pragma unroll
  for (int i = 0; i < 16; ++i) {
    const int c = half * 64 + i * 4;
    const float4 gg = *(const float4*)(g + c);
    const float4 bb = *(const float4*)(b + c);
    sN[wave][row * DPAIR + c + 0] = (bf16_t)((va[i].x - mean) * rstd * gg.x + bb.x);
    sN[wave][row * DPAIR + c + 1] = (bf16_t)((va[i].y - mean) * rstd * gg.y + bb.y);
    sN[wave][row * DPAIR + c + 2] = (bf16_t)((va[i].z - mean) * rstd * gg.z + bb.z);
    sN[wave][row * DPAIR + c + 3] = (bf16_t)((va[i].w - mean) * rstd * gg.w + bb.w);
  }
  __syncthreads();
  f32x8 acc = zero8();
#pragma unroll
  for (int cc = 0; cc < 4; ++cc) {
    bf16x16 af = frag_a(&sN[wave][(lane & 15) * DPAIR + cc * 32], lane);
    bf16x16 bf = frag_b(WbT + (size_t)(lane & 15) * DPAIR + cc * 32, lane);
    acc = wmma_bf16(af, bf, acc);
  }
  const int h = lane & 15;
  if (h < NH) {
#pragma unroll
    for (int r = 0; r < 8; ++r) {
      const int m = (lane < 16) ? r : r + 8;
      bias[((size_t)q * SEQ + (kb + m)) * NH + h] = acc[r];
    }
  }
}

// ---------------------------------------------------------------------------
// GEMM: C(M x N) = act(alpha * A(MxK bf16) @ BT^T + biasv) + resid, with
// f32 and/or bf16 (optionally transposed) outputs.  64x64 block tile, 8 waves
// (4x2 of 16x32), K in 32-steps; tiles staged to LDS by the TDM (double
// buffered) with s_wait_tensorcnt + barriers.
// ---------------------------------------------------------------------------
__global__ __launch_bounds__(256) void k_gemm(const bf16_t* __restrict__ A,
                                              const bf16_t* __restrict__ BT,
                                              int M, int N, int K, float alpha,
                                              const float* __restrict__ biasv, int act,
                                              const float* __restrict__ resid,
                                              float* __restrict__ outf,
                                              bf16_t* __restrict__ outh,
                                              int outh_trans, int outh_ld) {
  __shared__ bf16_t sA[2][64 * 32];
  __shared__ bf16_t sB[2][64 * 32];
  const int tid = threadIdx.x, lane = tid & 31, wave = tid >> 5;
  const int mw = wave >> 1, nw = wave & 1;
  const int bm = blockIdx.x * 64, bn = blockIdx.y * 64;
  f32x8 acc[2] = {zero8(), zero8()};
  const int nkt = K >> 5;
  int buf = 0;
  (void)buf;
#if USE_TDM
  if (wave == 0) {
    tdm_load_tile2d((unsigned int)(size_t)&sA[0][0], A + (size_t)bm * K, 32u, 64u, (unsigned int)K);
    tdm_load_tile2d((unsigned int)(size_t)&sB[0][0], BT + (size_t)bn * K, 32u, 64u, (unsigned int)K);
  }
  for (int kt = 0; kt < nkt; ++kt) {
    if (wave == 0) __builtin_amdgcn_s_wait_tensorcnt(0);
    __syncthreads();                       // tile[buf] visible to all waves
    if (wave == 0 && kt + 1 < nkt) {       // prefetch next tile into other buffer
      const int k2 = (kt + 1) << 5;
      tdm_load_tile2d((unsigned int)(size_t)&sA[buf ^ 1][0], A + (size_t)bm * K + k2, 32u, 64u, (unsigned int)K);
      tdm_load_tile2d((unsigned int)(size_t)&sB[buf ^ 1][0], BT + (size_t)bn * K + k2, 32u, 64u, (unsigned int)K);
    }
    bf16x16 af = frag_a(&sA[buf][(mw * 16 + (lane & 15)) * 32], lane);
#pragma unroll
    for (int sub = 0; sub < 2; ++sub) {
      bf16x16 bf = frag_b(&sB[buf][(nw * 32 + sub * 16 + (lane & 15)) * 32], lane);
      acc[sub] = wmma_bf16(af, bf, acc[sub]);
    }
    __syncthreads();
    buf ^= 1;
  }
#else
  for (int kt = 0; kt < nkt; ++kt) {
    const int k0 = kt << 5;
    {
      const int idx = tid * 8;             // 2048 halves per tile, 16B per thread
      const int r = idx >> 5, c = idx & 31;
      *(uint4*)&sA[0][idx] = *(const uint4*)(A + (size_t)(bm + r) * K + k0 + c);
      *(uint4*)&sB[0][idx] = *(const uint4*)(BT + (size_t)(bn + r) * K + k0 + c);
    }
    __syncthreads();
    bf16x16 af = frag_a(&sA[0][(mw * 16 + (lane & 15)) * 32], lane);
#pragma unroll
    for (int sub = 0; sub < 2; ++sub) {
      bf16x16 bf = frag_b(&sB[0][(nw * 32 + sub * 16 + (lane & 15)) * 32], lane);
      acc[sub] = wmma_bf16(af, bf, acc[sub]);
    }
    __syncthreads();
  }
#endif
#pragma unroll
  for (int sub = 0; sub < 2; ++sub) {
    const int cn = bn + nw * 32 + sub * 16 + (lane & 15);
#pragma unroll
    for (int r = 0; r < 8; ++r) {
      const int m = bm + mw * 16 + ((lane < 16) ? r : r + 8);
      float v = alpha * acc[sub][r];
      if (biasv) v += biasv[cn];
      if (act == 1) v = fmaxf(v, 0.f);
      else if (act == 2) v = 1.f / (1.f + __expf(-v));
      if (resid) v += resid[(size_t)m * N + cn];
      if (outf) outf[(size_t)m * N + cn] = v;
      if (outh) {
        if (outh_trans) outh[(size_t)cn * outh_ld + m] = (bf16_t)v;
        else            outh[(size_t)m * N + cn] = (bf16_t)v;
      }
    }
  }
}

// ---------------------------------------------------------------------------
// Attention: one block = 16 query rows x 1 head, full 1024-key pass.
// Phase 1: logits via wmma (K = DH = 32, one wmma per 16-key chunk) + pair bias
//          into LDS (64 KB).  Phase 2: softmax (shfl reductions) -> bf16 P.
// Phase 3: O = P @ V via wmma against transposed V, cross-wave LDS reduce.
// ---------------------------------------------------------------------------
__global__ __launch_bounds__(256) void k_attn(const bf16_t* __restrict__ Q,
                                              const bf16_t* __restrict__ Kmat,
                                              const bf16_t* __restrict__ VT,
                                              const float* __restrict__ bias,
                                              float* __restrict__ O) {
  __shared__ float  S[16 * SEQ];        // 64 KB logits
  __shared__ bf16_t P[16 * SEQ];        // 32 KB probabilities
  __shared__ float  Opart[8][16 * 32];  // 16 KB per-wave partials
  const int tid = threadIdx.x, lane = tid & 31, wave = tid >> 5;
  const int h = blockIdx.y;
  const int qb = blockIdx.x * 16;

  const bf16x16 qf = frag_a(Q + (size_t)(qb + (lane & 15)) * DMSA + h * DHH, lane);

  for (int c = wave; c < SEQ / 16; c += 8) {
    const int kb = c * 16;
    bf16x16 kf = frag_b(Kmat + (size_t)(kb + (lane & 15)) * DMSA + h * DHH, lane);
    f32x8 s = zero8();
    s = wmma_bf16(qf, kf, s);
    const int n = lane & 15;
#pragma unroll
    for (int r = 0; r < 8; ++r) {
      const int m = (lane < 16) ? r : r + 8;
      float v = s[r];
      if (bias) v += bias[((size_t)(qb + m) * SEQ + (kb + n)) * NH + h];
      S[m * SEQ + kb + n] = v;
    }
  }
  __syncthreads();

  for (int m = wave; m < 16; m += 8) {
    float mx = -3.4e38f;
    for (int j = lane; j < SEQ; j += 32) mx = fmaxf(mx, S[m * SEQ + j]);
#pragma unroll
    for (int o = 16; o > 0; o >>= 1) mx = fmaxf(mx, __shfl_xor(mx, o, 32));
    float sum = 0.f;
    for (int j = lane; j < SEQ; j += 32) {
      const float e = __expf(S[m * SEQ + j] - mx);
      S[m * SEQ + j] = e;
      sum += e;
    }
#pragma unroll
    for (int o = 16; o > 0; o >>= 1) sum += __shfl_xor(sum, o, 32);
    const float inv = 1.f / sum;
    for (int j = lane; j < SEQ; j += 32) P[m * SEQ + j] = (bf16_t)(S[m * SEQ + j] * inv);
  }
  __syncthreads();

  f32x8 acc[2] = {zero8(), zero8()};
  for (int c = wave; c < SEQ / 32; c += 8) {
    const int kb = c * 32;
    bf16x16 pf = frag_a(&P[(lane & 15) * SEQ + kb], lane);
#pragma unroll
    for (int sub = 0; sub < 2; ++sub) {
      const int dh = sub * 16 + (lane & 15);
      bf16x16 vf = frag_b(VT + (size_t)(h * DHH + dh) * SEQ + kb, lane);
      acc[sub] = wmma_bf16(pf, vf, acc[sub]);
    }
  }
#pragma unroll
  for (int sub = 0; sub < 2; ++sub)
#pragma unroll
    for (int r = 0; r < 8; ++r) {
      const int m = (lane < 16) ? r : r + 8;
      Opart[wave][m * 32 + sub * 16 + (lane & 15)] = acc[sub][r];
    }
  __syncthreads();
  for (int idx = tid; idx < 512; idx += 256) {
    const int m = idx >> 5, d = idx & 31;
    float s = 0.f;
#pragma unroll
    for (int w = 0; w < 8; ++w) s += Opart[w][m * 32 + d];
    O[(size_t)(qb + m) * DMSA + h * DHH + d] = s;
  }
}

// gated[i] = bf16( sigmoid_gate[i] * o[i] )  (gate already sigmoid'ed by GEMM)
__global__ __launch_bounds__(256) void k_gate(const float* __restrict__ g,
                                              const float* __restrict__ o,
                                              bf16_t* __restrict__ out) {
  const int i = blockIdx.x * 256 + threadIdx.x;
  out[i] = (bf16_t)(g[i] * o[i]);
}

// ---------------------------------------------------------------------------
extern "C" void kernel_launch(void* const* d_in, const int* in_sizes, int n_in,
                              void* d_out, int out_size, void* d_ws, size_t ws_size,
                              hipStream_t stream) {
  (void)in_sizes; (void)n_in; (void)out_size; (void)ws_size;
  const float* msa       = (const float*)d_in[0];
  const float* pair      = (const float*)d_in[1];
  const float* ln_node_g = (const float*)d_in[2];
  const float* ln_node_b = (const float*)d_in[3];
  const float* ln_pair_g = (const float*)d_in[4];
  const float* ln_pair_b = (const float*)d_in[5];
  const float* Ws[12] = {
    (const float*)d_in[6],  (const float*)d_in[7],  (const float*)d_in[8],   // row Wq Wk Wv
    (const float*)d_in[10], (const float*)d_in[12],                          // row Wg Wo
    (const float*)d_in[14], (const float*)d_in[15], (const float*)d_in[16],  // col Wq Wk Wv
    (const float*)d_in[17], (const float*)d_in[19],                          // col Wg Wo
    (const float*)d_in[23], (const float*)d_in[25]                           // ff W1 W2
  };
  const float* row_Wb = (const float*)d_in[9];
  const float* row_bg = (const float*)d_in[11];
  const float* row_bo = (const float*)d_in[13];
  const float* col_bg = (const float*)d_in[18];
  const float* col_bo = (const float*)d_in[20];
  const float* ff_ln_g = (const float*)d_in[21];
  const float* ff_ln_b = (const float*)d_in[22];
  const float* ff_b1   = (const float*)d_in[24];
  const float* ff_b2   = (const float*)d_in[26];

  char* base = (char*)d_ws;
  size_t off = 0;
  auto alloc = [&](size_t bytes) -> void* {
    void* p = base + off;
    off = (off + bytes + 255) & ~(size_t)255;
    return p;
  };
  const size_t XF = (size_t)SEQ * DMSA * 4, XH = (size_t)SEQ * DMSA * 2;
  float*  biasb = (float*)alloc((size_t)SEQ * SEQ * NH * 4);   // 32 MB
  float*  x0f = (float*)alloc(XF);
  float*  x1f = (float*)alloc(XF);
  float*  x2f = (float*)alloc(XF);
  float*  gbuf = (float*)alloc(XF);
  float*  obuf = (float*)alloc(XF);
  bf16_t* x0h = (bf16_t*)alloc(XH);
  bf16_t* x1h = (bf16_t*)alloc(XH);
  bf16_t* qh  = (bf16_t*)alloc(XH);
  bf16_t* kh  = (bf16_t*)alloc(XH);
  bf16_t* vth = (bf16_t*)alloc(XH);    // transposed (256 x 1024)
  bf16_t* gath = (bf16_t*)alloc(XH);
  bf16_t* hh  = (bf16_t*)alloc(XH);
  bf16_t* th  = (bf16_t*)alloc(XH);
  bf16_t* wT[12];
  for (int i = 0; i < 12; ++i) wT[i] = (bf16_t*)alloc((size_t)DMSA * DMSA * 2);
  bf16_t* wbT = (bf16_t*)alloc((size_t)16 * DPAIR * 2);

  const float SCALE_Q = 0.17677669529663688f;   // 1/sqrt(32)

  for (int i = 0; i < 12; ++i) k_wT<<<256, 256, 0, stream>>>(Ws[i], wT[i]);
  k_wbT<<<8, 256, 0, stream>>>(row_Wb, wbT);

  k_ln<<<SEQ, 256, 0, stream>>>(msa, ln_node_g, ln_node_b, x0f, x0h);
  k_pair_bias<<<dim3(SEQ, SEQ / 64), 128, 0, stream>>>(pair, ln_pair_g, ln_pair_b, wbT, biasb);

  auto gemm = [&](const bf16_t* A, const bf16_t* BT, float alpha, const float* bv,
                  int act, const float* res, float* of, bf16_t* oh, int tr, int ld) {
    k_gemm<<<dim3(SEQ / 64, DMSA / 64), 256, 0, stream>>>(A, BT, SEQ, DMSA, DMSA,
                                                          alpha, bv, act, res, of, oh, tr, ld);
  };

  // ---- row attention (with pair bias) ----
  gemm(x0h, wT[0], SCALE_Q, nullptr, 0, nullptr, nullptr, qh, 0, 0);
  gemm(x0h, wT[1], 1.f,     nullptr, 0, nullptr, nullptr, kh, 0, 0);
  gemm(x0h, wT[2], 1.f,     nullptr, 0, nullptr, nullptr, vth, 1, SEQ);
  gemm(x0h, wT[3], 1.f,     row_bg,  2, nullptr, gbuf, nullptr, 0, 0);
  k_attn<<<dim3(SEQ / 16, NH), 256, 0, stream>>>(qh, kh, vth, biasb, obuf);
  k_gate<<<SEQ * DMSA / 256, 256, 0, stream>>>(gbuf, obuf, gath);
  gemm(gath, wT[4], 1.f, row_bo, 0, x0f, x1f, x1h, 0, 0);

  // ---- col attention (no bias) ----
  gemm(x1h, wT[5], SCALE_Q, nullptr, 0, nullptr, nullptr, qh, 0, 0);
  gemm(x1h, wT[6], 1.f,     nullptr, 0, nullptr, nullptr, kh, 0, 0);
  gemm(x1h, wT[7], 1.f,     nullptr, 0, nullptr, nullptr, vth, 1, SEQ);
  gemm(x1h, wT[8], 1.f,     col_bg,  2, nullptr, gbuf, nullptr, 0, 0);
  k_attn<<<dim3(SEQ / 16, NH), 256, 0, stream>>>(qh, kh, vth, nullptr, obuf);
  k_gate<<<SEQ * DMSA / 256, 256, 0, stream>>>(gbuf, obuf, gath);
  gemm(gath, wT[9], 1.f, col_bo, 0, x1f, x2f, nullptr, 0, 0);

  // ---- feed-forward ----
  k_ln<<<SEQ, 256, 0, stream>>>(x2f, ff_ln_g, ff_ln_b, nullptr, hh);
  gemm(hh, wT[10], 1.f, ff_b1, 1, nullptr, nullptr, th, 0, 0);
  gemm(th, wT[11], 1.f, ff_b2, 0, x2f, (float*)d_out, nullptr, 0, 0);
}
